// PhaseTracker_16286515986774
// MI455X (gfx1250) — matrix operand order repaired and checked by reading, not compiled, not run on hardware
//
#include <hip/hip_runtime.h>
#include <math.h>

// ---------------- constants -------------------------------------------------
#define N_DET     4096
#define D_IN      128
#define H_DIM     64
#define N_OSC     28
#define K_PAD     64          // 28 cos + 28 sin, zero-padded to 64
#define N_STEPS   5
#define DT_F      0.01f
#define THRESH    0.3f
#define TWO_PI_F  6.2831853071795864769f

// denom = (sqrt(28)+1e-6)^2 computed in double, as in the reference
#define SIM_SCALE ((float)(1.0 / ((5.2915026221291814 + 1e-6) * (5.2915026221291814 + 1e-6))))

typedef _Float16 v16h __attribute__((ext_vector_type(16)));
typedef _Float16 v8h  __attribute__((ext_vector_type(8)));
typedef float    v8f  __attribute__((ext_vector_type(8)));

union ABFrag { v16h v; v8h h[2]; };

__device__ __forceinline__ float pymod2pi(float x) {
    float r = fmodf(x, TWO_PI_F);
    return (r < 0.0f) ? r + TWO_PI_F : r;
}

__device__ __forceinline__ float softplus_ref(float x) {
    // jax.nn.softplus: max(x,0) + log1p(exp(-|x|))
    return fmaxf(x, 0.0f) + log1pf(expf(-fabsf(x)));
}

__device__ __forceinline__ float omega_of(int o) {
    float hz = (o < 4) ? 2.0f : (o < 12) ? 6.0f : 40.0f;
    return TWO_PI_F * hz;
}

// ---------------- kernel 1: encode (+ optional evolve) ----------------------
// One thread per detection row. Writes F[row][0..27]=cos(ph), [28..55]=sin(ph),
// [56..63]=0 as f16, ready for the WMMA similarity GEMM.
__global__ void encode_evolve_kernel(const float* __restrict__ X,
                                     const float* __restrict__ W1, const float* __restrict__ b1,
                                     const float* __restrict__ W2, const float* __restrict__ b2,
                                     const float* __restrict__ Wa1, const float* __restrict__ ba1,
                                     const float* __restrict__ Wa2, const float* __restrict__ ba2,
                                     _Float16* __restrict__ F, int do_evolve) {
    int r = blockIdx.x * blockDim.x + threadIdx.x;
    if (r >= N_DET) return;
    const float* x = X + (size_t)r * D_IN;

    float h[H_DIM];
    float ph[N_OSC];

    // ---- phase MLP: h = relu(x@W1 + b1); ph = mod(h@W2 + b2, 2pi)
    #pragma unroll
    for (int j = 0; j < H_DIM; ++j) h[j] = b1[j];
    for (int k = 0; k < D_IN; ++k) {
        float xv = x[k];
        const float* w = W1 + (size_t)k * H_DIM;
        #pragma unroll
        for (int j = 0; j < H_DIM; ++j) h[j] = fmaf(xv, w[j], h[j]);
    }
    #pragma unroll
    for (int j = 0; j < H_DIM; ++j) h[j] = fmaxf(h[j], 0.0f);

    #pragma unroll
    for (int o = 0; o < N_OSC; ++o) ph[o] = b2[o];
    for (int k = 0; k < H_DIM; ++k) {
        float hv = h[k];
        const float* w = W2 + (size_t)k * N_OSC;
        #pragma unroll
        for (int o = 0; o < N_OSC; ++o) ph[o] = fmaf(hv, w[o], ph[o]);
    }
    #pragma unroll
    for (int o = 0; o < N_OSC; ++o) ph[o] = pymod2pi(ph[o]);

    if (do_evolve) {
        // ---- amplitude MLP (reuse h)
        float am[N_OSC];
        #pragma unroll
        for (int j = 0; j < H_DIM; ++j) h[j] = ba1[j];
        for (int k = 0; k < D_IN; ++k) {
            float xv = x[k];
            const float* w = Wa1 + (size_t)k * H_DIM;
            #pragma unroll
            for (int j = 0; j < H_DIM; ++j) h[j] = fmaf(xv, w[j], h[j]);
        }
        #pragma unroll
        for (int j = 0; j < H_DIM; ++j) h[j] = fmaxf(h[j], 0.0f);
        #pragma unroll
        for (int o = 0; o < N_OSC; ++o) am[o] = ba2[o];
        for (int k = 0; k < H_DIM; ++k) {
            float hv = h[k];
            const float* w = Wa2 + (size_t)k * N_OSC;
            #pragma unroll
            for (int o = 0; o < N_OSC; ++o) am[o] = fmaf(hv, w[o], am[o]);
        }
        #pragma unroll
        for (int o = 0; o < N_OSC; ++o) am[o] = softplus_ref(am[o]);

        // ---- Kuramoto mean-field, 5 steps
        for (int st = 0; st < N_STEPS; ++st) {
            float c[N_OSC], s[N_OSC];
            float mc = 0.0f, ms = 0.0f;
            #pragma unroll
            for (int o = 0; o < N_OSC; ++o) {
                float co, si;
                sincosf(ph[o], &si, &co);
                c[o] = co; s[o] = si;
                mc += co; ms += si;
            }
            mc *= (1.0f / (float)N_OSC);
            ms *= (1.0f / (float)N_OSC);
            #pragma unroll
            for (int o = 0; o < N_OSC; ++o) {
                float coupling = ms * c[o] - mc * s[o];
                ph[o] = pymod2pi(ph[o] + DT_F * (omega_of(o) + am[o] * coupling));
            }
        }
    }

    // ---- write f16 feature row: [cos | sin | 0-pad]
    _Float16* f = F + (size_t)r * K_PAD;
    #pragma unroll
    for (int o = 0; o < N_OSC; ++o) {
        float co, si;
        sincosf(ph[o], &si, &co);
        f[o]          = (_Float16)co;
        f[N_OSC + o]  = (_Float16)si;
    }
    #pragma unroll
    for (int o = 2 * N_OSC; o < K_PAD; ++o) f[o] = (_Float16)0.0f;
}

// ---------------- kernel 2: similarity GEMM via WMMA ------------------------
// sim = (Ft @ Ft1^T) * SIM_SCALE.  K = 64 -> two v_wmma_f32_16x16x32_f16.
// Block = 256 threads = 8 waves; each wave owns one 16x16 output tile,
// block covers a 32(M) x 64(N) region.  Grid = (4096/64, 4096/32).
__global__ void sim_wmma_kernel(const _Float16* __restrict__ Ft,
                                const _Float16* __restrict__ Ft1,
                                float* __restrict__ sim) {
    int wave = threadIdx.x >> 5;
    int lane = threadIdx.x & 31;
    int wm = wave >> 2;                 // 0..1
    int wn = wave & 3;                  // 0..3
    int tm = blockIdx.y * 2 + wm;       // 16-row tile index
    int tn = blockIdx.x * 4 + wn;       // 16-col tile index
    int g  = lane >> 4;                 // half-wave group
    int mn = lane & 15;                 // M (for A) / N (for B,C,D)

    const _Float16* Arow = Ft  + (size_t)(tm * 16 + mn) * K_PAD;
    const _Float16* Brow = Ft1 + (size_t)(tn * 16 + mn) * K_PAD;

    v8f c = {};
    #pragma unroll
    for (int kc = 0; kc < K_PAD; kc += 32) {
        ABFrag a, b;
        // A 16x32 fragment: halves 0..7 -> K = kc + 8g .. +7 ; halves 8..15 -> K = kc + 16 + 8g .. +7
        a.h[0] = *(const v8h*)(Arow + kc + 8 * g);
        a.h[1] = *(const v8h*)(Arow + kc + 16 + 8 * g);
        // B 32x16 fragment (column mn): halves 0..7 -> K = kc + 8g .. +7 ; halves 8..15 -> K = kc + 16 + 8g .. +7
        b.h[0] = *(const v8h*)(Brow + kc + 8 * g);
        b.h[1] = *(const v8h*)(Brow + kc + 16 + 8 * g);
        c = __builtin_amdgcn_wmma_f32_16x16x32_f16(
                /*neg_a=*/false, a.v, /*neg_b=*/false, b.v,
                /*c_mod=*/(short)0, c, /*reuse_a=*/false, /*reuse_b=*/false);
    }

    // C/D layout: VGPR r -> row = tm*16 + 8*g + r, col = tn*16 + mn
    float* out = sim + (size_t)(tm * 16 + 8 * g) * N_DET + (size_t)(tn * 16) + mn;
    #pragma unroll
    for (int r = 0; r < 8; ++r)
        out[(size_t)r * N_DET] = c[r] * SIM_SCALE;
}

// ---------------- kernel 3: per-row max / argmax (first-index tiebreak) -----
__global__ void rowmax_kernel(const float* __restrict__ sim,
                              float* __restrict__ max_sims,
                              int* __restrict__ max_idxs) {
    int row = blockIdx.x;
    int tid = threadIdx.x;
    const float* p = sim + (size_t)row * N_DET;

    float best = -INFINITY;
    int bi = 0;
    for (int j = tid; j < N_DET; j += 256) {
        float v = p[j];
        if (v > best) { best = v; bi = j; }   // per-thread j ascending -> keeps first max
    }
    __shared__ float sv[256];
    __shared__ int   si[256];
    sv[tid] = best; si[tid] = bi;
    __syncthreads();
    for (int off = 128; off > 0; off >>= 1) {
        if (tid < off) {
            float v2 = sv[tid + off]; int i2 = si[tid + off];
            if (v2 > sv[tid] || (v2 == sv[tid] && i2 < si[tid])) { sv[tid] = v2; si[tid] = i2; }
        }
        __syncthreads();
    }
    if (tid == 0) { max_sims[row] = sv[0]; max_idxs[row] = si[0]; }
}

// ---------------- kernel 4: stable descending argsort via O(N^2) rank -------
__global__ void rank_kernel(const float* __restrict__ ms, int* __restrict__ order) {
    int i = blockIdx.x * blockDim.x + threadIdx.x;
    if (i >= N_DET) return;
    float mi = ms[i];
    int cnt = 0;
    for (int j = 0; j < N_DET; ++j) {
        float mj = ms[j];
        cnt += (mj > mi) || ((mj == mi) && (j < i));   // stable: ties by index
    }
    order[cnt] = i;
}

// ---------------- kernel 5: sequential greedy matching ----------------------
__global__ void greedy_kernel(const int* __restrict__ order,
                              const int* __restrict__ max_idxs,
                              const float* __restrict__ max_sims,
                              int* __restrict__ matches,
                              int* __restrict__ used) {
    if (blockIdx.x == 0 && threadIdx.x == 0) {
        for (int j = 0; j < N_DET; ++j) used[j] = 0;
        for (int k = 0; k < N_DET; ++k) {
            int i = order[k];
            int j = max_idxs[i];
            bool ok = (used[j] == 0) && (max_sims[i] > THRESH);
            if (ok) used[j] = 1;
            matches[i] = ok ? j : -1;
        }
    }
}

// ---------------- launch ----------------------------------------------------
extern "C" void kernel_launch(void* const* d_in, const int* in_sizes, int n_in,
                              void* d_out, int out_size, void* d_ws, size_t ws_size,
                              hipStream_t stream) {
    (void)in_sizes; (void)n_in; (void)out_size; (void)ws_size;

    const float* det_t  = (const float*)d_in[0];
    const float* det_t1 = (const float*)d_in[1];
    const float* Wp1 = (const float*)d_in[2];
    const float* bp1 = (const float*)d_in[3];
    const float* Wp2 = (const float*)d_in[4];
    const float* bp2 = (const float*)d_in[5];
    const float* Wa1 = (const float*)d_in[6];
    const float* ba1 = (const float*)d_in[7];
    const float* Wa2 = (const float*)d_in[8];
    const float* ba2 = (const float*)d_in[9];

    // outputs: matches (4096 int32) then sim (4096*4096 f32), flat-concatenated
    int*   matches = (int*)d_out;
    float* sim     = (float*)d_out + N_DET;

    // workspace layout
    char* ws = (char*)d_ws;
    _Float16* F_t   = (_Float16*)(ws);                       // 512 KB
    _Float16* F_t1  = (_Float16*)(ws + 524288);              // 512 KB
    float*    msims = (float*)(ws + 1048576);                // 16 KB
    int*      midxs = (int*)  (ws + 1048576 + 16384);
    int*      order = (int*)  (ws + 1048576 + 2 * 16384);
    int*      used  = (int*)  (ws + 1048576 + 3 * 16384);

    // 1) encode (+evolve) both detection sets into f16 feature matrices
    encode_evolve_kernel<<<N_DET / 256, 256, 0, stream>>>(
        det_t, Wp1, bp1, Wp2, bp2, Wa1, ba1, Wa2, ba2, F_t, 1);
    encode_evolve_kernel<<<N_DET / 256, 256, 0, stream>>>(
        det_t1, Wp1, bp1, Wp2, bp2, Wa1, ba1, Wa2, ba2, F_t1, 0);

    // 2) similarity matrix via WMMA
    dim3 simGrid(N_DET / 64, N_DET / 32);
    sim_wmma_kernel<<<simGrid, 256, 0, stream>>>(F_t, F_t1, sim);

    // 3) per-row max/argmax
    rowmax_kernel<<<N_DET, 256, 0, stream>>>(sim, msims, midxs);

    // 4) stable descending rank -> order
    rank_kernel<<<N_DET / 256, 256, 0, stream>>>(msims, order);

    // 5) sequential greedy match
    greedy_kernel<<<1, 32, 0, stream>>>(order, midxs, msims, matches, used);
}
